// TransformerEncoder_34746285424677
// MI455X (gfx1250) — matrix operand order repaired and checked by reading, not compiled, not run on hardware
//
#include <hip/hip_runtime.h>

// ---------------------------------------------------------------------------
// Types for CDNA5 WMMA (wave32)
// ---------------------------------------------------------------------------
typedef __bf16 bf16;
typedef __attribute__((ext_vector_type(16))) __bf16 v16bf;
typedef __attribute__((ext_vector_type(8)))  __bf16 v8bf;
typedef __attribute__((ext_vector_type(8)))  float  v8f;

#define BB   8
#define NN   1024
#define KNN  20
#define NH   8
#define LQ   8
#define DH   16
#define MROWS (BB * NN)          // 8192
#define EDGE_CNT (float)(BB * NN * KNN)  // 163840

__device__ __forceinline__ float lrelu(float v) { return v > 0.f ? v : 0.2f * v; }

// ---------------------------------------------------------------------------
// Register-blocked WMMA GEMM:  C[M,Nc] = A[M,Kc] @ W[Nc,Kc]^T (+bias, leaky)
// One wave per 64x64 output block = 4x4 tiles of 16x16.
// Per k-step: 4 B-fragments loaded once, A-fragments streamed one at a time
// (live set: 16 acc + 4 B + 1 A ~= 180 VGPRs; __launch_bounds__(32,1) lifts
// the allocator cap so accumulators stay resident -- no scratch spills).
// Requires M % 64 == 0, Nc % 64 == 0, Kc % 32 == 0 (true for all call sites).
// A lane/K swizzle per CDNA5 ISA 7.12.2 (16-bit A 16x32); B per SWMMAC layout.
// ---------------------------------------------------------------------------
__global__ void __launch_bounds__(32, 1)
k_gemm(const bf16* __restrict__ A, const bf16* __restrict__ W,
       const float* __restrict__ bias, float* __restrict__ C,
       int M, int Kc, int Nc, int act) {
    int lane = threadIdx.x;
    int mlow = lane & 15;
    int half = lane >> 4;          // 0 or 1
    int m0 = blockIdx.y * 64;
    int n0 = blockIdx.x * 64;
    const bf16* arow[4];
    const bf16* brow[4];
#pragma unroll
    for (int t = 0; t < 4; ++t) {
        arow[t] = A + (size_t)(m0 + t * 16 + mlow) * Kc;
        brow[t] = W + (size_t)(n0 + t * 16 + mlow) * Kc;
    }
    v8f acc[16];
#pragma unroll
    for (int t = 0; t < 16; ++t) acc[t] = (v8f){};
    int aoff = 8 * half;           // A: lane<16 -> K {0..7,16..23}; lane>=16 +8
    int boff = 8 * half;           // B: lane<16 -> K {0..7,16..23}; lane>=16 +8
#pragma unroll 1
    for (int k0 = 0; k0 < Kc; k0 += 32) {
        v16bf bfr[4];
#pragma unroll
        for (int t = 0; t < 4; ++t) {
            v8bf b0 = *(const v8bf*)(brow[t] + k0 + boff);
            v8bf b1 = *(const v8bf*)(brow[t] + k0 + 16 + boff);
#pragma unroll
            for (int e = 0; e < 8; ++e) { bfr[t][e] = b0[e]; bfr[t][e + 8] = b1[e]; }
        }
#pragma unroll
        for (int i = 0; i < 4; ++i) {
            v8bf a0 = *(const v8bf*)(arow[i] + k0 + aoff);
            v8bf a1 = *(const v8bf*)(arow[i] + k0 + 16 + aoff);
            v16bf af;
#pragma unroll
            for (int e = 0; e < 8; ++e) { af[e] = a0[e]; af[e + 8] = a1[e]; }
#pragma unroll
            for (int j = 0; j < 4; ++j)
                acc[i * 4 + j] = __builtin_amdgcn_wmma_f32_16x16x32_bf16(
                    false, af, false, bfr[j], (short)0, acc[i * 4 + j], false, false);
        }
    }
#pragma unroll
    for (int i = 0; i < 4; ++i) {
#pragma unroll
        for (int j = 0; j < 4; ++j) {
            int col = n0 + j * 16 + mlow;
            float bv = bias ? bias[col] : 0.f;
#pragma unroll
            for (int jj = 0; jj < 8; ++jj) {
                int r = m0 + i * 16 + jj + 8 * half;  // C layout: VGPR jj -> M=jj(+8)
                float v = acc[i * 4 + j][jj] + bv;
                if (act) v = lrelu(v);
                C[(size_t)r * Nc + col] = v;
            }
        }
    }
}

// ---------------------------------------------------------------------------
// f32 -> bf16 conversion
// ---------------------------------------------------------------------------
__global__ void k_cvt(const float* __restrict__ in, bf16* __restrict__ out, size_t n) {
    size_t i = (size_t)blockIdx.x * blockDim.x + threadIdx.x;
    if (i < n) out[i] = (bf16)in[i];
}

// ---------------------------------------------------------------------------
// kNN: one thread per point, top-20 smallest squared distances (incl. self)
// ---------------------------------------------------------------------------
__global__ void k_knn(const float* __restrict__ x, int* __restrict__ idx) {
    int t = blockIdx.x * blockDim.x + threadIdx.x;
    if (t >= MROWS) return;
    int b = t >> 10, n = t & (NN - 1);
    const float* xb = x + (size_t)b * NN * 3;
    float cx = xb[n * 3 + 0], cy = xb[n * 3 + 1], cz = xb[n * 3 + 2];
    float bd[KNN]; int bi[KNN];
#pragma unroll
    for (int i = 0; i < KNN; ++i) { bd[i] = 3.4e38f; bi[i] = 0; }
    for (int m = 0; m < NN; ++m) {
        float dx = xb[m * 3 + 0] - cx, dy = xb[m * 3 + 1] - cy, dz = xb[m * 3 + 2] - cz;
        float d = dx * dx + dy * dy + dz * dz;
        if (d < bd[KNN - 1]) {
            int p = KNN - 1;
            while (p > 0 && bd[p - 1] > d) { bd[p] = bd[p - 1]; bi[p] = bi[p - 1]; --p; }
            bd[p] = d; bi[p] = m;
        }
    }
    for (int i = 0; i < KNN; ++i) idx[(size_t)t * KNN + i] = bi[i];
}

// ---------------------------------------------------------------------------
// Edge-feature moments: S1[6] and upper-tri S2[21] of e = [neigh-ctr, ctr]
// (exact BN stats of the linear edge conv derive from these). Deterministic.
// ---------------------------------------------------------------------------
__global__ void k_edge_moments(const float* __restrict__ x, const int* __restrict__ idx,
                               float* __restrict__ moments) {
    __shared__ float sh[256 * 27];
    int tid = threadIdx.x;
    float loc[27];
#pragma unroll
    for (int j = 0; j < 27; ++j) loc[j] = 0.f;
    for (int bn = tid; bn < MROWS; bn += 256) {
        int b = bn >> 10, n = bn & (NN - 1);
        const float* xb = x + (size_t)b * NN * 3;
        float e[6];
        e[3] = xb[n * 3 + 0]; e[4] = xb[n * 3 + 1]; e[5] = xb[n * 3 + 2];
        for (int k = 0; k < KNN; ++k) {
            int m = idx[(size_t)bn * KNN + k];
            e[0] = xb[m * 3 + 0] - e[3];
            e[1] = xb[m * 3 + 1] - e[4];
            e[2] = xb[m * 3 + 2] - e[5];
            int t = 6;
#pragma unroll
            for (int i = 0; i < 6; ++i) {
                loc[i] += e[i];
#pragma unroll
                for (int j2 = 0; j2 < 6; ++j2)
                    if (j2 >= i) loc[t++] += e[i] * e[j2];
            }
        }
    }
    for (int j = 0; j < 27; ++j) sh[tid * 27 + j] = loc[j];
    __syncthreads();
    for (int s = 128; s > 0; s >>= 1) {
        if (tid < s)
            for (int j = 0; j < 27; ++j) sh[tid * 27 + j] += sh[(tid + s) * 27 + j];
        __syncthreads();
    }
    if (tid == 0)
        for (int j = 0; j < 27; ++j) moments[j] = sh[j];
}

// ---------------------------------------------------------------------------
// Edge conv + exact BN + leaky + max over K  ->  ycat[:, 0:128]
// block = 128 threads (one per out-channel), grid = B*N
// ---------------------------------------------------------------------------
__global__ void k_edge_apply(const float* __restrict__ x, const int* __restrict__ idx,
                             const float* __restrict__ ew, const float* __restrict__ eb,
                             const float* __restrict__ moments,
                             const float* __restrict__ g, const float* __restrict__ beta,
                             float* __restrict__ ycat) {
    __shared__ float se[KNN][6];
    __shared__ float sw[128 * 6];
    int bn = blockIdx.x;
    int b = bn >> 10, n = bn & (NN - 1);
    int o = threadIdx.x;
    for (int i = o; i < 128 * 6; i += 128) sw[i] = ew[i];
    const float* xb = x + (size_t)b * NN * 3;
    float cx = xb[n * 3 + 0], cy = xb[n * 3 + 1], cz = xb[n * 3 + 2];
    if (o < KNN) {
        int m = idx[(size_t)bn * KNN + o];
        se[o][0] = xb[m * 3 + 0] - cx; se[o][1] = xb[m * 3 + 1] - cy;
        se[o][2] = xb[m * 3 + 2] - cz;
        se[o][3] = cx; se[o][4] = cy; se[o][5] = cz;
    }
    __syncthreads();
    float wo[6];
#pragma unroll
    for (int c = 0; c < 6; ++c) wo[c] = sw[o * 6 + c];
    float bias = eb[o];
    float s1w = 0.f;
#pragma unroll
    for (int i = 0; i < 6; ++i) s1w += wo[i] * moments[i];
    float q2 = 0.f; int t = 6;
#pragma unroll
    for (int i = 0; i < 6; ++i)
#pragma unroll
        for (int j2 = i; j2 < 6; ++j2) {
            float v = moments[t++];
            q2 += wo[i] * wo[j2] * v * ((i == j2) ? 1.f : 2.f);
        }
    float mh  = s1w / EDGE_CNT + bias;
    float eh2 = q2 / EDGE_CNT + 2.f * bias * (s1w / EDGE_CNT) + bias * bias;
    float rstd = rsqrtf(eh2 - mh * mh + 1e-5f);
    float gg = g[o], bb2 = beta[o];
    float mv = -3.4e38f;
    for (int k = 0; k < KNN; ++k) {
        float h = bias;
#pragma unroll
        for (int c = 0; c < 6; ++c) h += se[k][c] * wo[c];
        mv = fmaxf(mv, lrelu((h - mh) * rstd * gg + bb2));
    }
    ycat[(size_t)bn * 1024 + o] = mv;
}

// ---------------------------------------------------------------------------
// Graph max pooling: pooled[bn, c] = max_k ycat[b*N + idx[bn,k], off + c]
// ---------------------------------------------------------------------------
__global__ void k_gather_max(const float* __restrict__ ycat, const int* __restrict__ idx,
                             float* __restrict__ pooled, int off, int Cin) {
    __shared__ int sidx[KNN];
    int bn = blockIdx.x;
    int b = bn >> 10;
    if (threadIdx.x < KNN) sidx[threadIdx.x] = idx[(size_t)bn * KNN + threadIdx.x];
    __syncthreads();
    for (int c = threadIdx.x; c < Cin; c += blockDim.x) {
        float mv = -3.4e38f;
        for (int k = 0; k < KNN; ++k)
            mv = fmaxf(mv, ycat[((size_t)(b * NN + sidx[k])) * 1024 + off + c]);
        pooled[(size_t)bn * Cin + c] = mv;
    }
}

// ---------------------------------------------------------------------------
// BN stats (deterministic): one thread per channel, full-M reduction
// ---------------------------------------------------------------------------
__global__ void k_bn_stats(const float* __restrict__ z, float* __restrict__ mean,
                           float* __restrict__ var, int M, int C) {
    int c = blockIdx.x * blockDim.x + threadIdx.x;
    if (c >= C) return;
    float s = 0.f, s2 = 0.f;
    for (int r = 0; r < M; ++r) {
        float v = z[(size_t)r * C + c];
        s += v; s2 += v * v;
    }
    float m = s / (float)M;
    mean[c] = m;
    var[c] = s2 / (float)M - m * m;
}

__global__ void k_bn_apply(const float* __restrict__ z, const float* __restrict__ mean,
                           const float* __restrict__ var, const float* __restrict__ g,
                           const float* __restrict__ be, float* __restrict__ out,
                           int M, int C, int ostride, int ooff, int act) {
    size_t i = (size_t)blockIdx.x * blockDim.x + threadIdx.x;
    if (i >= (size_t)M * C) return;
    int r = (int)(i / C), c = (int)(i % C);
    float v = (z[i] - mean[c]) * rsqrtf(var[c] + 1e-5f) * g[c] + be[c];
    if (act) v = lrelu(v);
    out[(size_t)r * ostride + ooff + c] = v;
}

// ---------------------------------------------------------------------------
// Column max over points: xmax[b, c] = max_n z[b*N+n, c]   (C = 1024)
// ---------------------------------------------------------------------------
__global__ void k_colmax(const float* __restrict__ z, float* __restrict__ xmax) {
    int b = blockIdx.y;
    int c = blockIdx.x * blockDim.x + threadIdx.x;
    float mv = -3.4e38f;
    for (int n = 0; n < NN; ++n)
        mv = fmaxf(mv, z[((size_t)(b * NN + n)) * 1024 + c]);
    xmax[b * 1024 + c] = mv;
}

// ---------------------------------------------------------------------------
// memory = leaky(x @ proji_w^T + b); writes f32 and bf16
// ---------------------------------------------------------------------------
__global__ void k_memproj(const float* __restrict__ x, const float* __restrict__ pw,
                          const float* __restrict__ pb, float* __restrict__ mem,
                          bf16* __restrict__ mem_bf) {
    int bn = blockIdx.x;
    int o = threadIdx.x;
    const float* xr = x + (size_t)bn * 3;
    float v = pb[o] + xr[0] * pw[o * 3] + xr[1] * pw[o * 3 + 1] + xr[2] * pw[o * 3 + 2];
    v = lrelu(v);
    mem[(size_t)bn * 128 + o] = v;
    mem_bf[(size_t)bn * 128 + o] = (bf16)v;
}

// ---------------------------------------------------------------------------
// Multi-head attention, online softmax, one thread per (b, h, q).
// Q: (B*LQ,128)  K,V: (B*Lk,128)  O: (B*LQ,128); dh=16, scale=1/4
// ---------------------------------------------------------------------------
__global__ void k_attn(const float* __restrict__ Q, const float* __restrict__ Kx,
                       const float* __restrict__ V, float* __restrict__ O, int Lk) {
    int t = blockIdx.x * blockDim.x + threadIdx.x;
    if (t >= BB * NH * LQ) return;
    int b = t / (NH * LQ);
    int h = (t / LQ) % NH;
    int qi = t % LQ;
    const float* qp = Q + ((size_t)(b * LQ + qi)) * 128 + h * DH;
    float qv[DH], acc[DH];
#pragma unroll
    for (int d = 0; d < DH; ++d) { qv[d] = qp[d]; acc[d] = 0.f; }
    float m = -3.4e38f, l = 0.f;
    const float scale = 0.25f;
    for (int k = 0; k < Lk; ++k) {
        const float* kp = Kx + ((size_t)(b * Lk + k)) * 128 + h * DH;
        float s = 0.f;
#pragma unroll
        for (int d = 0; d < DH; ++d) s += qv[d] * kp[d];
        s *= scale;
        float nm = fmaxf(m, s);
        float f = __expf(m - nm);
        float p = __expf(s - nm);
        l = l * f + p;
        const float* vp = V + ((size_t)(b * Lk + k)) * 128 + h * DH;
#pragma unroll
        for (int d = 0; d < DH; ++d) acc[d] = acc[d] * f + p * vp[d];
        m = nm;
    }
    float* op = O + ((size_t)(b * LQ + qi)) * 128 + h * DH;
    float rl = 1.f / l;
#pragma unroll
    for (int d = 0; d < DH; ++d) op[d] = acc[d] * rl;
}

// ---------------------------------------------------------------------------
// out = LayerNorm(qin + delta) over last dim 128. delta may be null.
// grid = rows, block = 128. Safe in-place.
// ---------------------------------------------------------------------------
__global__ void k_res_ln(const float* __restrict__ qin, const float* __restrict__ delta,
                         const float* __restrict__ g, const float* __restrict__ bb,
                         float* __restrict__ out) {
    __shared__ float red[128];
    int r = blockIdx.x, c = threadIdx.x;
    float v = qin[(size_t)r * 128 + c] + (delta ? delta[(size_t)r * 128 + c] : 0.f);
    red[c] = v; __syncthreads();
    for (int s = 64; s > 0; s >>= 1) { if (c < s) red[c] += red[c + s]; __syncthreads(); }
    float mean = red[0] * (1.f / 128.f);
    __syncthreads();
    float dv = v - mean;
    red[c] = dv * dv; __syncthreads();
    for (int s = 64; s > 0; s >>= 1) { if (c < s) red[c] += red[c + s]; __syncthreads(); }
    float var = red[0] * (1.f / 128.f);
    out[(size_t)r * 128 + c] = dv * rsqrtf(var + 1e-5f) * g[c] + bb[c];
}

// ---------------------------------------------------------------------------
// Host orchestration
// ---------------------------------------------------------------------------
extern "C" void kernel_launch(void* const* d_in, const int* in_sizes, int n_in,
                              void* d_out, int out_size, void* d_ws, size_t ws_size,
                              hipStream_t stream) {
    const float* x       = (const float*)d_in[0];
    const float* edge_w  = (const float*)d_in[1];
    const float* edge_b  = (const float*)d_in[2];
    const float* edge_g  = (const float*)d_in[3];
    const float* edge_be = (const float*)d_in[4];
    const float* pc_w[3]  = {(const float*)d_in[5],  (const float*)d_in[9],  (const float*)d_in[13]};
    const float* pc_b[3]  = {(const float*)d_in[6],  (const float*)d_in[10], (const float*)d_in[14]};
    const float* pc_g[3]  = {(const float*)d_in[7],  (const float*)d_in[11], (const float*)d_in[15]};
    const float* pc_be[3] = {(const float*)d_in[8],  (const float*)d_in[12], (const float*)d_in[16]};
    const float* fc_w  = (const float*)d_in[17];
    const float* fc_b  = (const float*)d_in[18];
    const float* fc_g  = (const float*)d_in[19];
    const float* fc_be = (const float*)d_in[20];
    const float* projc_w = (const float*)d_in[21];
    const float* projc_b = (const float*)d_in[22];
    const float* proji_w = (const float*)d_in[23];
    const float* proji_b = (const float*)d_in[24];
    const float* sa_w = (const float*)d_in[25];
    const float* sa_b = (const float*)d_in[26];
    const float* ca_w = (const float*)d_in[27];
    const float* ca_b = (const float*)d_in[28];
    const float* ff_w1 = (const float*)d_in[29];
    const float* ff_b1 = (const float*)d_in[30];
    const float* ff_w2 = (const float*)d_in[31];
    const float* ff_b2 = (const float*)d_in[32];
    const float* ln_g = (const float*)d_in[33];
    const float* ln_b = (const float*)d_in[34];
    const float* fln_g = (const float*)d_in[35];
    const float* fln_b = (const float*)d_in[36];

    const int M = MROWS;
    char* base = (char*)d_ws;
    size_t woff = 0;
    auto alloc = [&](size_t bytes) -> void* {
        void* p = base + woff;
        woff += (bytes + 255) & ~(size_t)255;
        return p;
    };
    int*   idx       = (int*)  alloc((size_t)M * KNN * 4);
    float* ycat      = (float*)alloc((size_t)M * 1024 * 4);
    bf16*  ycat_bf   = (bf16*) alloc((size_t)M * 1024 * 2);
    float* pooled    = (float*)alloc((size_t)M * 512 * 4);
    bf16*  pooled_bf = (bf16*) alloc((size_t)M * 512 * 2);
    float* z         = (float*)alloc((size_t)M * 1024 * 4);
    float* mean      = (float*)alloc(1024 * 4);
    float* var       = (float*)alloc(1024 * 4);
    float* moments   = (float*)alloc(128 * 4);
    float* xmax      = (float*)alloc(8192 * 4);
    float* mem       = (float*)alloc((size_t)M * 128 * 4);
    bf16*  mem_bf    = (bf16*) alloc((size_t)M * 128 * 2);
    float* kbuf      = (float*)alloc((size_t)M * 128 * 4);
    float* vbuf      = (float*)alloc((size_t)M * 128 * 4);
    float* q         = (float*)alloc(64 * 128 * 4);
    float* qh        = (float*)alloc(64 * 128 * 4);
    float* khs       = (float*)alloc(64 * 128 * 4);
    float* vhs       = (float*)alloc(64 * 128 * 4);
    float* attn_o    = (float*)alloc(64 * 128 * 4);
    float* delta     = (float*)alloc(64 * 128 * 4);
    float* h1        = (float*)alloc(64 * 512 * 4);
    bf16*  sbf       = (bf16*) alloc(64 * 512 * 2);
    bf16*  w_pc[3];
    w_pc[0] = (bf16*)alloc(128 * 128 * 2);
    w_pc[1] = (bf16*)alloc(256 * 128 * 2);
    w_pc[2] = (bf16*)alloc(512 * 256 * 2);
    bf16* w_fc    = (bf16*)alloc((size_t)1024 * 1024 * 2);
    bf16* w_projc = (bf16*)alloc(128 * 128 * 2);
    bf16* w_sa    = (bf16*)alloc((size_t)4 * 4 * 128 * 128 * 2);
    bf16* w_ca    = (bf16*)alloc((size_t)4 * 4 * 128 * 128 * 2);
    bf16* w_f1    = (bf16*)alloc((size_t)4 * 512 * 128 * 2);
    bf16* w_f2    = (bf16*)alloc((size_t)4 * 128 * 512 * 2);

    auto cvt = [&](const float* in, bf16* out, size_t n) {
        k_cvt<<<dim3((unsigned)((n + 255) / 256)), dim3(256), 0, stream>>>(in, out, n);
    };
    auto gemm = [&](const bf16* A, const bf16* W, const float* bias, float* C,
                    int Mm, int Kc, int Nc, int act) {
        k_gemm<<<dim3(Nc / 64, Mm / 64), dim3(32), 0, stream>>>(A, W, bias, C, Mm, Kc, Nc, act);
    };

    // weight conversions (f32 -> bf16)
    cvt(pc_w[0], w_pc[0], 128 * 128);
    cvt(pc_w[1], w_pc[1], 256 * 128);
    cvt(pc_w[2], w_pc[2], 512 * 256);
    cvt(fc_w, w_fc, (size_t)1024 * 1024);
    cvt(projc_w, w_projc, 128 * 128);
    cvt(sa_w, w_sa, (size_t)4 * 4 * 128 * 128);
    cvt(ca_w, w_ca, (size_t)4 * 4 * 128 * 128);
    cvt(ff_w1, w_f1, (size_t)4 * 512 * 128);
    cvt(ff_w2, w_f2, (size_t)4 * 128 * 512);

    // kNN + edge conv (exact analytic BN) -> ycat[:, 0:128]
    k_knn<<<dim3(M / 64), dim3(64), 0, stream>>>(x, idx);
    k_edge_moments<<<dim3(1), dim3(256), 0, stream>>>(x, idx, moments);
    k_edge_apply<<<dim3(M), dim3(128), 0, stream>>>(x, idx, edge_w, edge_b, moments,
                                                    edge_g, edge_be, ycat);

    // three graph-pool + 1x1 conv + BN + act layers
    const int cin[3]  = {128, 128, 256};
    const int cout[3] = {128, 256, 512};
    const int ioff[3] = {0, 128, 256};
    const int ooff[3] = {128, 256, 512};
    for (int i = 0; i < 3; ++i) {
        k_gather_max<<<dim3(M), dim3(128), 0, stream>>>(ycat, idx, pooled, ioff[i], cin[i]);
        cvt(pooled, pooled_bf, (size_t)M * cin[i]);
        gemm(pooled_bf, w_pc[i], pc_b[i], z, M, cin[i], cout[i], 0);
        k_bn_stats<<<dim3((cout[i] + 255) / 256), dim3(256), 0, stream>>>(z, mean, var, M, cout[i]);
        k_bn_apply<<<dim3((unsigned)(((size_t)M * cout[i] + 255) / 256)), dim3(256), 0, stream>>>(
            z, mean, var, pc_g[i], pc_be[i], ycat, M, cout[i], 1024, ooff[i], 1);
    }

    // fc 1024 -> 1024 + BN + act (in place on z), then global max over points
    cvt(ycat, ycat_bf, (size_t)M * 1024);
    gemm(ycat_bf, w_fc, fc_b, z, M, 1024, 1024, 0);
    k_bn_stats<<<dim3(4), dim3(256), 0, stream>>>(z, mean, var, M, 1024);
    k_bn_apply<<<dim3((unsigned)(((size_t)M * 1024 + 255) / 256)), dim3(256), 0, stream>>>(
        z, mean, var, fc_g, fc_be, z, M, 1024, 1024, 0, 1);
    k_colmax<<<dim3(4, BB), dim3(256), 0, stream>>>(z, xmax);

    // queries (B*8, 128) and memory (B*N, 128)
    cvt(xmax, sbf, 8192);
    gemm(sbf, w_projc, projc_b, q, 64, 128, 128, 1);
    k_memproj<<<dim3(M), dim3(128), 0, stream>>>(x, proji_w, proji_b, mem, mem_bf);

    // 4 transformer decoder layers
    for (int l = 0; l < 4; ++l) {
        const bf16* sw  = w_sa + (size_t)l * 4 * 16384;
        const bf16* cw  = w_ca + (size_t)l * 4 * 16384;
        const float* sb = sa_b + l * 4 * 128;
        const float* cb = ca_b + l * 4 * 128;
        // self-attention
        cvt(q, sbf, 64 * 128);
        gemm(sbf, sw + 0 * 16384, sb + 0,   qh,  64, 128, 128, 0);
        gemm(sbf, sw + 1 * 16384, sb + 128, khs, 64, 128, 128, 0);
        gemm(sbf, sw + 2 * 16384, sb + 256, vhs, 64, 128, 128, 0);
        k_attn<<<dim3(8), dim3(64), 0, stream>>>(qh, khs, vhs, attn_o, LQ);
        cvt(attn_o, sbf, 64 * 128);
        gemm(sbf, sw + 3 * 16384, sb + 384, delta, 64, 128, 128, 0);
        k_res_ln<<<dim3(64), dim3(128), 0, stream>>>(q, delta, ln_g + (l * 3 + 0) * 128,
                                                     ln_b + (l * 3 + 0) * 128, q);
        // cross-attention
        cvt(q, sbf, 64 * 128);
        gemm(sbf, cw + 0 * 16384, cb + 0, qh, 64, 128, 128, 0);
        gemm(mem_bf, cw + 1 * 16384, cb + 128, kbuf, M, 128, 128, 0);
        gemm(mem_bf, cw + 2 * 16384, cb + 256, vbuf, M, 128, 128, 0);
        k_attn<<<dim3(8), dim3(64), 0, stream>>>(qh, kbuf, vbuf, attn_o, NN);
        cvt(attn_o, sbf, 64 * 128);
        gemm(sbf, cw + 3 * 16384, cb + 384, delta, 64, 128, 128, 0);
        k_res_ln<<<dim3(64), dim3(128), 0, stream>>>(q, delta, ln_g + (l * 3 + 1) * 128,
                                                     ln_b + (l * 3 + 1) * 128, q);
        // feed-forward
        cvt(q, sbf, 64 * 128);
        gemm(sbf, w_f1 + (size_t)l * 512 * 128, ff_b1 + l * 512, h1, 64, 128, 512, 1);
        cvt(h1, sbf, 64 * 512);
        gemm(sbf, w_f2 + (size_t)l * 128 * 512, ff_b2 + l * 128, delta, 64, 512, 128, 0);
        k_res_ln<<<dim3(64), dim3(128), 0, stream>>>(q, delta, ln_g + (l * 3 + 2) * 128,
                                                     ln_b + (l * 3 + 2) * 128, q);
    }

    // final LayerNorm straight into d_out (B, 1024) == (64 rows, 128)
    k_res_ln<<<dim3(64), dim3(128), 0, stream>>>(q, nullptr, fln_g, fln_b, (float*)d_out);
}
